// DissectSpatial_16569983828166
// MI455X (gfx1250) — compile-verified, matrix operand order checked by reading, not run on hardware
//
#include <hip/hip_runtime.h>
#include <hip/hip_fp16.h>

// Problem constants (match the reference)
#define NN   50000
#define NPAD 50048    // NN rounded up to multiple of BM: OOB rows are load-safe garbage
#define EE   800000
#define DIN  500
#define D0   502      // DIN + D_POS
#define D0P  512      // D0 padded to multiple of BK (zero-filled)
#define LAT  128
#define CC   10
#define NEG_SLOPE 0.2f

typedef __attribute__((ext_vector_type(16))) _Float16 v16h;
typedef __attribute__((ext_vector_type(8)))  _Float16 v8h;
typedef __attribute__((ext_vector_type(8)))  float    v8f;

// ---------------- helpers ----------------
__device__ __forceinline__ int f2ord(float f) {
    int i = __float_as_int(f);
    return i >= 0 ? i : (i ^ 0x7FFFFFFF);   // monotonic int ordering of floats
}
__device__ __forceinline__ float ord2f(int i) {
    int j = i >= 0 ? i : (i ^ 0x7FFFFFFF);
    return __int_as_float(j);
}

// ---------------- elementwise kernels ----------------
// xin_h[N,512] = concat(x, pos) in f16, cols 502..511 zero
__global__ void k_build_xin(const float* __restrict__ x, const float* __restrict__ pos,
                            _Float16* __restrict__ xin, int n) {
    long long i = (long long)blockIdx.x * blockDim.x + threadIdx.x;
    long long total = (long long)n * D0P;
    if (i >= total) return;
    int r = (int)(i >> 9), c = (int)(i & (D0P - 1));
    float v = 0.f;
    if (c < DIN)      v = x[(long long)r * DIN + c];
    else if (c < D0)  v = pos[(long long)r * 2 + (c - DIN)];
    xin[i] = (_Float16)v;
}

// dst[RD,Ncol] f16 from src[RS,Ncol] f32, rows >= RS zero-filled
__global__ void k_cvt_pad(const float* __restrict__ src, _Float16* __restrict__ dst,
                          int rs, int rd, int ncol) {
    int i = blockIdx.x * blockDim.x + threadIdx.x;
    if (i >= rd * ncol) return;
    int r = i / ncol;
    dst[i] = (r < rs) ? (_Float16)src[i] : (_Float16)0.f;
}

__global__ void k_gat_init(float* __restrict__ denom, float* __restrict__ numer,
                           int* __restrict__ smax, int n) {
    int i = blockIdx.x * blockDim.x + threadIdx.x;
    if (i < n) { denom[i] = 0.f; smax[i] = f2ord(-__builtin_inff()); }
    if (i < n * LAT) numer[i] = 0.f;
}

// ---------------- WMMA GEMM: C = act(A[f16] @ B[f16] + bias) ----------------
// BM=128, BN=64, BK=32; 256 threads = 8 waves, wave w owns rows [w*16,w*16+16)
// x 64 cols (4 wmma tiles sharing one A fragment).
// REQUIRES: K % 32 == 0, lda/ldb % 8 == 0, A readable for BM-rounded rows.
#define BM 128
#define BN 64
#define BK 32
#define LDS_STRIDE 40   // halfs; 80B rows keep all v8h chunks 16B-aligned

__global__ __launch_bounds__(256) void k_wmma_gemm(
    const _Float16* __restrict__ A, int lda,
    const _Float16* __restrict__ B, int ldb,
    const float* __restrict__ bias,
    float* __restrict__ Cf, _Float16* __restrict__ Ch, int ldc,
    int M, int K, int relu)
{
    __shared__ __align__(16) _Float16 As[BM][LDS_STRIDE];   // [row][k]
    __shared__ __align__(16) _Float16 Bs[BN][LDS_STRIDE];   // transposed: [col][k]

    const int tid  = threadIdx.x;
    const int wave = tid >> 5;
    const int lane = tid & 31;
    const int m    = lane & 15;      // row (A) / col (B,C) within 16x16 tile
    const int hi   = lane >> 4;      // K-half select per CDNA5 16-bit operand layout
    const int rowBase = blockIdx.y * BM;
    const int colBase = blockIdx.x * BN;

    // A staging map: 128 rows x 4 chunks of 8 halfs = 512 chunks; 2 per thread
    const int ar0 = tid >> 2;            // rows ar0 and ar0+64
    const int aq  = (tid & 3) * 8;       // 8-half chunk within the 32-wide k tile
    const _Float16* pA0 = A + (size_t)(rowBase + ar0) * lda + aq;
    const _Float16* pA1 = A + (size_t)(rowBase + ar0 + 64) * lda + aq;
    // B staging map: 32 k-rows x 8 chunks of 8 cols = 256 chunks; 1 per thread
    const int bk = tid >> 3;             // k within tile
    const int bc = (tid & 7) * 8;        // col chunk
    const _Float16* pB = B + (size_t)bk * ldb + colBase + bc;

    v8f acc[4];
    #pragma unroll
    for (int t = 0; t < 4; ++t)
        #pragma unroll
        for (int j = 0; j < 8; ++j) acc[t][j] = 0.f;

    // prologue: first tile into registers
    v8h a0 = *(const v8h*)pA0;
    v8h a1 = *(const v8h*)pA1;
    v8h bb = *(const v8h*)pB;

    for (int k0 = 0; k0 < K; k0 += BK) {
        // registers -> LDS
        *(v8h*)&As[ar0][aq]      = a0;
        *(v8h*)&As[ar0 + 64][aq] = a1;
        #pragma unroll
        for (int i = 0; i < 8; ++i) Bs[bc + i][bk] = bb[i];   // transpose scatter
        __syncthreads();

        // pipeline: issue next tile's global loads before the matrix math
        if (k0 + BK < K) {
            a0 = *(const v8h*)(pA0 + (k0 + BK));
            a1 = *(const v8h*)(pA1 + (k0 + BK));
            bb = *(const v8h*)(pB + (size_t)(k0 + BK) * ldb);
            if (k0 + 2 * BK < K) {
                __builtin_prefetch(pA0 + (k0 + 2 * BK), 0, 1);
                __builtin_prefetch(pA1 + (k0 + 2 * BK), 0, 1);
            }
        }

        // A fragment: lane (hi,m) holds K = hi*8..+7 and 16+hi*8..+7 (two b128s)
        const int arow = wave * 16 + m;
        v8h alo = *(const v8h*)&As[arow][hi * 8];
        v8h ahi = *(const v8h*)&As[arow][16 + hi * 8];
        v16h af = __builtin_shufflevector(alo, ahi, 0,1,2,3,4,5,6,7,8,9,10,11,12,13,14,15);

        #pragma unroll
        for (int nt = 0; nt < 4; ++nt) {
            const int bcol = nt * 16 + m;
            v8h blo = *(const v8h*)&Bs[bcol][hi * 8];
            v8h bhi = *(const v8h*)&Bs[bcol][16 + hi * 8];
            v16h bf = __builtin_shufflevector(blo, bhi, 0,1,2,3,4,5,6,7,8,9,10,11,12,13,14,15);
            acc[nt] = __builtin_amdgcn_wmma_f32_16x16x32_f16(
                false, af, false, bf, (short)0, acc[nt], false, false);
        }
        __syncthreads();
    }

    // C layout: VGPR v -> row = v + 8*hi within tile; col = lane&15
    #pragma unroll
    for (int nt = 0; nt < 4; ++nt) {
        const int c = colBase + nt * 16 + m;
        const float bcv = bias[c];
        #pragma unroll
        for (int v = 0; v < 8; ++v) {
            int r = rowBase + wave * 16 + hi * 8 + v;
            if (r < M) {
                float val = acc[nt][v] + bcv;
                if (relu) val = val > 0.f ? val : 0.f;
                if (Cf) Cf[(size_t)r * ldc + c] = val;
                if (Ch) Ch[(size_t)r * ldc + c] = (_Float16)val;
            }
        }
    }
}

// ---------------- GATv2 edge kernels (one wave per edge) ----------------
__global__ void k_edge_logits(const int* __restrict__ ei, const float* __restrict__ ea,
                              const float* __restrict__ xl, const float* __restrict__ xr,
                              const float* __restrict__ We, const float* __restrict__ att,
                              float* __restrict__ s_out, int* __restrict__ smax_ord, int E_) {
    int gid = blockIdx.x * blockDim.x + threadIdx.x;
    int e = gid >> 5, lane = gid & 31;
    if (e >= E_) return;
    int src = ei[e], dst = ei[E_ + e];
    float a = ea[e];
    float4 l  = ((const float4*)(xl + (size_t)src * LAT))[lane];
    float4 r  = ((const float4*)(xr + (size_t)dst * LAT))[lane];
    float4 w  = ((const float4*)We)[lane];
    float4 at = ((const float4*)att)[lane];
    float mv[4] = { l.x + r.x + a * w.x, l.y + r.y + a * w.y,
                    l.z + r.z + a * w.z, l.w + r.w + a * w.w };
    float av[4] = { at.x, at.y, at.z, at.w };
    float partial = 0.f;
    #pragma unroll
    for (int i = 0; i < 4; ++i) {
        float mm = mv[i] > 0.f ? mv[i] : NEG_SLOPE * mv[i];  // leaky relu
        partial += mm * av[i];
    }
    #pragma unroll
    for (int off = 16; off > 0; off >>= 1) partial += __shfl_down(partial, off, 32);
    if (lane == 0) {
        s_out[e] = partial;
        atomicMax(&smax_ord[dst], f2ord(partial));
    }
}

__global__ void k_edge_accum(const int* __restrict__ ei, const float* __restrict__ xl,
                             const float* __restrict__ s_in, const int* __restrict__ smax_ord,
                             float* __restrict__ denom, float* __restrict__ numer, int E_) {
    int gid = blockIdx.x * blockDim.x + threadIdx.x;
    int e = gid >> 5, lane = gid & 31;
    if (e >= E_) return;
    int src = ei[e], dst = ei[E_ + e];
    float ex = __expf(s_in[e] - ord2f(smax_ord[dst]));
    if (lane == 0) atomicAdd(&denom[dst], ex);
    float4 l = ((const float4*)(xl + (size_t)src * LAT))[lane];
    float* pn = numer + (size_t)dst * LAT + lane * 4;
    atomicAdd(&pn[0], ex * l.x);
    atomicAdd(&pn[1], ex * l.y);
    atomicAdd(&pn[2], ex * l.z);
    atomicAdd(&pn[3], ex * l.w);
}

// z = relu(numer/denom + bg), stored f16 for the decoder GEMM
__global__ void k_gat_finalize(const float* __restrict__ numer, const float* __restrict__ denom,
                               const float* __restrict__ bg, _Float16* __restrict__ zh, int n) {
    int i = blockIdx.x * blockDim.x + threadIdx.x;
    if (i >= n * LAT) return;
    int node = i >> 7, c = i & 127;
    float den = denom[node];
    float v = (den > 0.f) ? (numer[i] / den) : 0.f;
    v += bg[c];
    v = v > 0.f ? v : 0.f;
    zh[i] = (_Float16)v;
}

// logits = d1 @ Wd2 + bd2, softmax
__global__ void k_decoder(const float* __restrict__ d1, const float* __restrict__ Wd2,
                          const float* __restrict__ bd2, float* __restrict__ out, int n) {
    int i = blockIdx.x * blockDim.x + threadIdx.x;
    if (i >= n) return;
    float logit[CC];
    #pragma unroll
    for (int c = 0; c < CC; ++c) logit[c] = bd2[c];
    const float* row = d1 + (size_t)i * 64;
    for (int d = 0; d < 64; ++d) {
        float v = row[d];
        #pragma unroll
        for (int c = 0; c < CC; ++c) logit[c] += v * Wd2[d * CC + c];
    }
    float mx = logit[0];
    #pragma unroll
    for (int c = 1; c < CC; ++c) mx = fmaxf(mx, logit[c]);
    float sum = 0.f, ex[CC];
    #pragma unroll
    for (int c = 0; c < CC; ++c) { ex[c] = __expf(logit[c] - mx); sum += ex[c]; }
    float inv = 1.f / sum;
    #pragma unroll
    for (int c = 0; c < CC; ++c) out[(size_t)i * CC + c] = ex[c] * inv;
}

// ---------------- host side ----------------
extern "C" void kernel_launch(void* const* d_in, const int* in_sizes, int n_in,
                              void* d_out, int out_size, void* d_ws, size_t ws_size,
                              hipStream_t stream) {
    const float* x    = (const float*)d_in[0];
    const float* pos  = (const float*)d_in[1];
    const int*   ei   = (const int*)  d_in[2];
    const float* ea   = (const float*)d_in[3];
    const float* W1   = (const float*)d_in[4];
    const float* b1   = (const float*)d_in[5];
    const float* W2   = (const float*)d_in[6];
    const float* b2   = (const float*)d_in[7];
    const float* W3   = (const float*)d_in[8];
    const float* b3   = (const float*)d_in[9];
    const float* Wl   = (const float*)d_in[10];
    const float* bl   = (const float*)d_in[11];
    const float* Wr   = (const float*)d_in[12];
    const float* br   = (const float*)d_in[13];
    const float* We   = (const float*)d_in[14];
    const float* att  = (const float*)d_in[15];
    const float* bg   = (const float*)d_in[16];
    const float* Wd1  = (const float*)d_in[17];
    const float* bd1  = (const float*)d_in[18];
    const float* Wd2  = (const float*)d_in[19];
    const float* bd2  = (const float*)d_in[20];
    float* out = (float*)d_out;

    char* ws = (char*)d_ws;
    size_t off = 0;
    auto alloc = [&](size_t bytes) -> void* {
        void* p = ws + off;
        off = (off + bytes + 255) & ~(size_t)255;
        return p;
    };

    // f16 buffers padded to NPAD rows so tile loads never go out of bounds
    _Float16* xin_h = (_Float16*)alloc((size_t)NPAD * D0P * 2);
    _Float16* W1h   = (_Float16*)alloc((size_t)D0P * 512 * 2);   // rows 502..511 zero
    _Float16* W2h   = (_Float16*)alloc((size_t)512 * 256 * 2);
    _Float16* W3h   = (_Float16*)alloc((size_t)256 * LAT * 2);
    _Float16* Wlh   = (_Float16*)alloc((size_t)LAT * LAT * 2);
    _Float16* Wrh   = (_Float16*)alloc((size_t)LAT * LAT * 2);
    _Float16* Wd1h  = (_Float16*)alloc((size_t)LAT * 64 * 2);
    _Float16* H1h   = (_Float16*)alloc((size_t)NPAD * 512 * 2);
    _Float16* H2h   = (_Float16*)alloc((size_t)NPAD * 256 * 2);
    _Float16* H3h   = (_Float16*)alloc((size_t)NPAD * LAT * 2);
    float*    xl    = (float*)   alloc((size_t)NN * LAT * 4);
    float*    xr    = (float*)   alloc((size_t)NN * LAT * 4);
    float*    s_e   = (float*)   alloc((size_t)EE * 4);
    int*      smax  = (int*)     alloc((size_t)NN * 4);
    float*    denom = (float*)   alloc((size_t)NN * 4);
    float*    numer = (float*)   alloc((size_t)NN * LAT * 4);
    _Float16* zh    = (_Float16*)alloc((size_t)NPAD * LAT * 2);
    float*    d1    = (float*)   alloc((size_t)NPAD * 64 * 4);
    (void)ws_size; (void)n_in; (void)in_sizes; (void)out_size;

    const int T = 256;
    // Precision staging: f32 -> f16 once (weights + padded concat input)
    k_build_xin<<<(int)(((size_t)NN * D0P + T - 1) / T), T, 0, stream>>>(x, pos, xin_h, NN);
    k_cvt_pad<<<(D0P * 512 + T - 1) / T, T, 0, stream>>>(W1, W1h, D0, D0P, 512);
    k_cvt_pad<<<(512 * 256 + T - 1) / T, T, 0, stream>>>(W2, W2h, 512, 512, 256);
    k_cvt_pad<<<(256 * LAT + T - 1) / T, T, 0, stream>>>(W3, W3h, 256, 256, LAT);
    k_cvt_pad<<<(LAT * LAT + T - 1) / T, T, 0, stream>>>(Wl, Wlh, LAT, LAT, LAT);
    k_cvt_pad<<<(LAT * LAT + T - 1) / T, T, 0, stream>>>(Wr, Wrh, LAT, LAT, LAT);
    k_cvt_pad<<<(LAT * 64 + T - 1) / T, T, 0, stream>>>(Wd1, Wd1h, LAT, LAT, 64);
    k_gat_init<<<(NN * LAT + T - 1) / T, T, 0, stream>>>(denom, numer, smax, NN);

    const int MB = NPAD / BM;   // 391 row-blocks
    // Encoder MLP (WMMA): (502->)512->512->256->128
    k_wmma_gemm<<<dim3(512 / BN, MB), T, 0, stream>>>(xin_h, D0P, W1h, 512, b1,
                                                      nullptr, H1h, 512, NN, D0P, 1);
    k_wmma_gemm<<<dim3(256 / BN, MB), T, 0, stream>>>(H1h, 512, W2h, 256, b2,
                                                      nullptr, H2h, 256, NN, 512, 1);
    k_wmma_gemm<<<dim3(LAT / BN, MB), T, 0, stream>>>(H2h, 256, W3h, LAT, b3,
                                                      nullptr, H3h, LAT, NN, 256, 0);
    // GATv2 node transforms (WMMA): xl = h@Wl+bl, xr = h@Wr+br (f32 for edge math)
    k_wmma_gemm<<<dim3(LAT / BN, MB), T, 0, stream>>>(H3h, LAT, Wlh, LAT, bl,
                                                      xl, nullptr, LAT, NN, LAT, 0);
    k_wmma_gemm<<<dim3(LAT / BN, MB), T, 0, stream>>>(H3h, LAT, Wrh, LAT, br,
                                                      xr, nullptr, LAT, NN, LAT, 0);
    // Edge softmax: out[dst] = (sum_e exp(s-smax)*xl[src]) / (sum_e exp(s-smax))
    const int EB = (EE + 7) / 8;         // 8 edges (waves) per 256-thread block
    k_edge_logits<<<EB, T, 0, stream>>>(ei, ea, xl, xr, We, att, s_e, smax, EE);
    k_edge_accum <<<EB, T, 0, stream>>>(ei, xl, s_e, smax, denom, numer, EE);
    k_gat_finalize<<<(NN * LAT + T - 1) / T, T, 0, stream>>>(numer, denom, bg, zh, NN);
    // Decoder: 128->64 (WMMA) -> 64->10 + softmax (scalar)
    k_wmma_gemm<<<dim3(64 / BN, MB), T, 0, stream>>>(zh, LAT, Wd1h, 64, bd1,
                                                     d1, nullptr, 64, NN, LAT, 1);
    k_decoder<<<(NN + T - 1) / T, T, 0, stream>>>(d1, Wd2, bd2, out, NN);
}
